// DSVM_7430293422451
// MI455X (gfx1250) — compile-verified, matrix operand order checked by reading, not run on hardware
//
#include <hip/hip_runtime.h>
#include <hip/hip_bf16.h>
#include <math.h>

// ---------------------------------------------------------------------------
// Deep Kalman Filter VAE for MI455X (gfx1250, wave32, WMMA).
//
// All GEMMs run on v_wmma_f32_16x16x32_f16 (f32 accumulate). Both operands of
// every WMMA are pre-swizzled f16 fragment buffers (16 contiguous halves per
// lane -> two b128 loads per fragment per K-step, zero inner-loop converts):
//   lanes 0-15 : row = lane,    K in {0..7, 16..23} of the 32-wide slab
//   lanes 16-31: row = lane-16, K in {8..15, 24..31}
// Producer kernels write activations directly in fragment layout (f16) in
// addition to any f32 copies that are true outputs of the reference.
// K-loops are software-pipelined: fragments for kt+1 are loaded before the
// WMMA for kt issues, so operand-load latency overlaps matrix math.
// out_back lives only as f16 fragments (64 MB, resident in the 192 MB L2).
// Workspace requirement: ~85 MB.
// ---------------------------------------------------------------------------

typedef __attribute__((ext_vector_type(16))) _Float16 v16h;
typedef __attribute__((ext_vector_type(8)))  float    v8f;

#define T_DIM 512
#define B_DIM 128
#define H_DIM 512
#define Z_DIM 64
#define Y_DIM 32
#define LOG2PI 1.8378770664093453f

// ---- fragment helpers ------------------------------------------------------

// Load a pre-swizzled f16 fragment (A or B): contiguous 16 halves per lane.
__device__ inline v16h load_frag_sw(const _Float16* __restrict__ w,
                                    int KT, int rt, int kt, int lane) {
  return *(const v16h*)(w + ((size_t)((rt * KT + kt) * 32 + lane) << 4));
}

// Inverse mapping: (row, col) element -> index in fragment buffer with K-tiles
// count KT (= K/32). Matches load_frag_sw / dsvm_swizzle_kernel layout.
__device__ inline int frag_index(int row, int col, int KT) {
  int k   = col & 31;
  int grp = (k >> 3) & 1;                       // lanes 16-31 hold K+8 halves
  int e   = (k & 7) + ((k & 16) ? 8 : 0);       // element within the 16 halves
  return ((((row >> 4) * KT + (col >> 5)) * 32) + (row & 15) + 16 * grp) * 16 + e;
}

__device__ inline float softplusf(float x) {
  return fmaxf(x, 0.f) + log1pf(expf(-fabsf(x)));
}
__device__ inline float sigmoidf(float x) { return 1.f / (1.f + expf(-x)); }

// Software-pipelined fragment GEMM accumulation loop.
__device__ inline v8f gemm_frags(const _Float16* __restrict__ Xf, int rtA,
                                 const _Float16* __restrict__ Wf, int rtB,
                                 int KT, int lane, v8f c) {
  v16h a = load_frag_sw(Xf, KT, rtA, 0, lane);
  v16h b = load_frag_sw(Wf, KT, rtB, 0, lane);
  for (int kt = 0; kt < KT; ++kt) {
    v16h an = a, bn = b;
    if (kt + 1 < KT) {
      an = load_frag_sw(Xf, KT, rtA, kt + 1, lane);
      bn = load_frag_sw(Wf, KT, rtB, kt + 1, lane);
    }
    if (kt + 2 < KT) {
      __builtin_prefetch(Wf + ((size_t)((rtB * KT + kt + 2) * 32 + lane) << 4), 0, 1);
      __builtin_prefetch(Xf + ((size_t)((rtA * KT + kt + 2) * 32 + lane) << 4), 0, 1);
    }
    c = __builtin_amdgcn_wmma_f32_16x16x32_f16(false, a, false, b,
                                               (short)0, c, false, false);
    a = an; b = bn;
  }
  return c;
}

// ---- prep: f32 weight [O,K] (lead dim ldsrc, column offset col0) -> f16
// fragment layout -------------------------------------------------------------
__global__ void dsvm_swizzle_kernel(const float* __restrict__ src, int ldsrc,
                                    int col0, int O, int K,
                                    _Float16* __restrict__ dst) {
  int i = blockIdx.x * blockDim.x + threadIdx.x;   // one per (nt,kt,lane)
  int total = (O >> 4) * (K >> 5) * 32;
  if (i >= total) return;
  int lane = i & 31, tile = i >> 5;
  int KT = K >> 5;
  int kt = tile % KT, nt = tile / KT;
  int grp = lane >> 4, r = lane & 15;
  const float* p = src + (size_t)(nt * 16 + r) * ldsrc + col0 + kt * 32 + 8 * grp;
  _Float16* q = dst + (size_t)i * 16;
#pragma unroll
  for (int e = 0; e < 8; ++e)  q[e] = (_Float16)p[e];
#pragma unroll
  for (int e = 8; e < 16; ++e) q[e] = (_Float16)p[e + 8];
}

// ---- pack all y[t] tiles into fragment layout once (K = 32, KT = 1) --------
__global__ void dsvm_pack_y_kernel(const float* __restrict__ y,
                                   _Float16* __restrict__ yf) {
  int i = blockIdx.x * blockDim.x + threadIdx.x;   // T*B*Y elements
  if (i >= T_DIM * B_DIM * Y_DIM) return;
  int col = i & 31;
  int b   = (i >> 5) & (B_DIM - 1);
  int t   = i >> 12;
  yf[(size_t)t * (B_DIM * Y_DIM) + frag_index(b, col, 1)] = (_Float16)y[i];
}

// ---- zero init -------------------------------------------------------------
__global__ void dsvm_zero_kernel(float* __restrict__ w, int n,
                                 float* __restrict__ accs) {
  int i = blockIdx.x * blockDim.x + threadIdx.x;
  if (i < n) w[i] = 0.f;
  if (i < 2) accs[i] = 0.f;   // kld, nll accumulators in d_out[0..1]
}

// ---- generic WMMA linear: act(X1@W1^T [+ X2@W2^T] + b), M = 128 ------------
// Inputs are f16 fragment buffers. Outputs: optional f32 matrix (+copy) and
// optional f16 fragment buffer (requires N % 32 == 0).
// grid = (8 * N/16)/4 blocks of 128 threads (4 waves, 1 tile per wave)
__global__ void dsvm_linear_kernel(const _Float16* __restrict__ X1, int K1,
                                   const _Float16* __restrict__ W1,
                                   const _Float16* __restrict__ X2, int K2,
                                   const _Float16* __restrict__ W2,
                                   const float* __restrict__ bias, int N,
                                   int act, float* __restrict__ out,
                                   float* __restrict__ out2,
                                   _Float16* __restrict__ outf) {
  int lane = threadIdx.x & 31;
  int wv   = threadIdx.x >> 5;
  int Ntiles = N >> 4;
  int tile = blockIdx.x * 4 + wv;
  int mt = tile / Ntiles, nt = tile % Ntiles;

  v8f c = {};
  c = gemm_frags(X1, mt, W1, nt, K1 >> 5, lane, c);
  if (K2 > 0)
    c = gemm_frags(X2, mt, W2, nt, K2 >> 5, lane, c);

  int grp = lane >> 4;
  int cn  = lane & 15;
  int col = nt * 16 + cn;
  float bv = bias ? bias[col] : 0.f;
  int KTo = N >> 5;
#pragma unroll
  for (int i = 0; i < 8; ++i) {
    int row = mt * 16 + i + 8 * grp;   // C layout: VGPR i -> M=i (+8 hi half)
    float v = c[i] + bv;
    if (act == 1)      v = fmaxf(v, 0.f);
    else if (act == 2) v = softplusf(v);
    if (out)  out[(size_t)row * N + col] = v;
    if (out2) out2[(size_t)row * N + col] = v;
    if (outf) outf[frag_index(row, col, KTo)] = (_Float16)v;
  }
}

// ---- fused GRU layer step: Hnew = GRUCell(X, Hprev) ------------------------
// grid = 8 mtiles * 32 ntiles = 256 blocks, 192 threads (6 waves:
// waves 0-2 -> gh gates r,z,n over Hf/Whh (K=512)
// waves 3-5 -> gi gates r,z,n over Xf/Wih (K=Kx)), combine via LDS.
__global__ void dsvm_gru_kernel(const _Float16* __restrict__ Xf, int Kx,
                                const _Float16* __restrict__ Hf,
                                const float* __restrict__ Hprev,
                                const _Float16* __restrict__ Wih,
                                const _Float16* __restrict__ Whh,
                                const float* __restrict__ bih,
                                const float* __restrict__ bhh,
                                float* __restrict__ Hnew,
                                _Float16* __restrict__ Hnewf,
                                float* __restrict__ Hout2,
                                _Float16* __restrict__ Hout2f) {
  __shared__ float acc[2][3][16][16];   // [src: 0=gh 1=gi][gate][m][n]
  int tid = threadIdx.x, lane = tid & 31, wv = tid >> 5;
  int mt = blockIdx.x >> 5;   // 0..7
  int nt = blockIdx.x & 31;   // 0..31

  {
    int g = wv % 3, src = wv / 3;
    const _Float16* Xs = src ? Xf  : Hf;
    const _Float16* Ws = src ? Wih : Whh;
    int K  = src ? Kx : H_DIM;
    int gnt = g * 32 + nt;    // gate tile within 1536/16 = 96 column tiles
    v8f c = {};
    c = gemm_frags(Xs, mt, Ws, gnt, K >> 5, lane, c);
    int grp = lane >> 4, cn = lane & 15;
#pragma unroll
    for (int i = 0; i < 8; ++i) acc[src][g][i + 8 * grp][cn] = c[i];
  }
  __syncthreads();

  for (int idx = tid; idx < 256; idx += 192) {
    int m = idx >> 4, n = idx & 15;
    int col = nt * 16 + n;                    // 0..511
    float gir = acc[1][0][m][n] + bih[col];
    float ghr = acc[0][0][m][n] + bhh[col];
    float giz = acc[1][1][m][n] + bih[512 + col];
    float ghz = acc[0][1][m][n] + bhh[512 + col];
    float gin = acc[1][2][m][n] + bih[1024 + col];
    float ghn = acc[0][2][m][n] + bhh[1024 + col];
    float r  = sigmoidf(gir + ghr);
    float z  = sigmoidf(giz + ghz);
    float nn = tanhf(gin + r * ghn);
    int row = mt * 16 + m;
    float h  = Hprev[(size_t)row * H_DIM + col];
    float hv = (1.f - z) * nn + z * h;
    Hnew[(size_t)row * H_DIM + col] = hv;
    int fi = frag_index(row, col, H_DIM >> 5);
    Hnewf[fi] = (_Float16)hv;
    if (Hout2)  Hout2[(size_t)row * H_DIM + col] = hv;
    if (Hout2f) Hout2f[fi] = (_Float16)hv;
  }
}

// ---- sampling + KLD + x_in assembly (per step), 32 blocks x 256 ------------
__global__ void dsvm_sample_kld_kernel(const float* __restrict__ eps_t,
                                       const float* __restrict__ pos_m,
                                       const float* __restrict__ pos_s,
                                       const float* __restrict__ pri_m,
                                       const float* __restrict__ pri_s,
                                       const float* __restrict__ y_prev,
                                       const float* __restrict__ ystd,
                                       float* __restrict__ z_out,
                                       _Float16* __restrict__ zf,
                                       _Float16* __restrict__ xf,
                                       float* __restrict__ kld_acc) {
  __shared__ float red[256];
  int idx = blockIdx.x * 256 + threadIdx.x;   // B*Z = 8192
  int b = idx >> 6, j = idx & 63;
  float pm = pos_m[idx], ps = pos_s[idx];
  float qm = pri_m[idx], qs = pri_s[idx];
  float z = eps_t[idx] * ps + pm;
  z_out[idx] = z;
  zf[frag_index(b, j, 2)] = (_Float16)z;            // z fragment (K=64)
  xf[frag_index(b, 32 + j, 4)] = (_Float16)z;       // x_in fragment (K=128)
  if (j < 32) {
    float yp = y_prev ? y_prev[b * 32 + j] : 0.f;
    xf[frag_index(b, j, 4)]      = (_Float16)yp;
    xf[frag_index(b, 96 + j, 4)] = (_Float16)ystd[b * 32 + j];
  }
  float d = pm - qm;
  float term = 0.5f * (2.f * logf(qs) - 2.f * logf(ps) +
                       (ps * ps + d * d) / (qs * qs) - 1.f);
  red[threadIdx.x] = term;
  __syncthreads();
  for (int s = 128; s > 0; s >>= 1) {
    if (threadIdx.x < s) red[threadIdx.x] += red[threadIdx.x + s];
    __syncthreads();
  }
  if (threadIdx.x == 0) atomicAdd(kld_acc, red[0]);
}

// ---- NLL accumulation (per step), 16 blocks x 256 --------------------------
__global__ void dsvm_nll_kernel(const float* __restrict__ y_t,
                                const float* __restrict__ ye_s,
                                float* __restrict__ nll_acc) {
  __shared__ float red[256];
  int idx = blockIdx.x * 256 + threadIdx.x;   // B*Y = 4096
  float s = ye_s[idx], yv = y_t[idx];
  red[threadIdx.x] = 0.5f * LOG2PI + logf(s) + yv * yv / (2.f * s * s);
  __syncthreads();
  for (int st = 128; st > 0; st >>= 1) {
    if (threadIdx.x < st) red[threadIdx.x] += red[threadIdx.x + st];
    __syncthreads();
  }
  if (threadIdx.x == 0) atomicAdd(nll_acc, red[0]);
}

// ---------------------------------------------------------------------------
extern "C" void kernel_launch(void* const* d_in, const int* in_sizes, int n_in,
                              void* d_out, int out_size, void* d_ws,
                              size_t ws_size, hipStream_t stream) {
  (void)in_sizes; (void)n_in; (void)out_size; (void)ws_size;
  const float* y      = (const float*)d_in[0];
  const float* eps    = (const float*)d_in[1];
  const float* ztr_W  = (const float*)d_in[2];
  const float* ztr_b  = (const float*)d_in[3];
  const float* ztr_mW = (const float*)d_in[4];
  const float* ztr_mb = (const float*)d_in[5];
  const float* ztr_sW = (const float*)d_in[6];
  const float* ztr_sb = (const float*)d_in[7];
  const float* yem_W  = (const float*)d_in[8];
  const float* yem_b  = (const float*)d_in[9];
  const float* yem_sW = (const float*)d_in[10];
  const float* yem_sb = (const float*)d_in[11];
  const float* zp_W1  = (const float*)d_in[12];
  const float* zp_b1  = (const float*)d_in[13];
  const float* zp_W2  = (const float*)d_in[14];
  const float* zp_b2  = (const float*)d_in[15];
  const float* zp_mW  = (const float*)d_in[16];
  const float* zp_mb  = (const float*)d_in[17];
  const float* zp_sW  = (const float*)d_in[18];
  const float* zp_sb  = (const float*)d_in[19];
  const float* fWih0  = (const float*)d_in[20];
  const float* fWhh0  = (const float*)d_in[21];
  const float* fbih0  = (const float*)d_in[22];
  const float* fbhh0  = (const float*)d_in[23];
  const float* fWih1  = (const float*)d_in[24];
  const float* fWhh1  = (const float*)d_in[25];
  const float* fbih1  = (const float*)d_in[26];
  const float* fbhh1  = (const float*)d_in[27];
  const float* bWih0  = (const float*)d_in[28];
  const float* bWhh0  = (const float*)d_in[29];
  const float* bbih0  = (const float*)d_in[30];
  const float* bbhh0  = (const float*)d_in[31];
  const float* bWih1  = (const float*)d_in[32];
  const float* bWhh1  = (const float*)d_in[33];
  const float* bbih1  = (const float*)d_in[34];
  const float* bbhh1  = (const float*)d_in[35];

  float* out = (float*)d_out;
  float* ws  = (float*)d_ws;

  // ---- workspace layout (float offsets; frag regions are f16 halves) ----
  const size_t BH  = (size_t)B_DIM * H_DIM;    // 65536 elements
  const size_t OFF_H32   = 0;                  // 8 f32 H buffers (ping-pong)
  const size_t OFF_YSTD  = OFF_H32 + 8 * BH;   // 524288
  const size_t OFF_HFRAG = OFF_YSTD + (size_t)B_DIM * Y_DIM;  // 528384
  const size_t OFF_ZFRAG = OFF_HFRAG + 4 * BH; // 8 H frag bufs = 8*BH halves
  const size_t ZERO_N    = OFF_ZFRAG + ((size_t)B_DIM * Z_DIM) / 2; // 794624
  const size_t OFF_XINF  = ZERO_N;                       // 16384 h = 8192 f
  const size_t OFF_ZTHF  = OFF_XINF + 8192;
  const size_t OFF_Q1F   = OFF_ZTHF + 8192;
  const size_t OFF_Q2F   = OFF_Q1F + 8192;
  const size_t OFF_YEHF  = OFF_Q2F + 8192;               // 8192 h = 4096 f
  const size_t OFF_YFRG  = OFF_YEHF + 4096;              // T*B*Y halves
  const size_t OFF_OBUF  = OFF_YFRG + ((size_t)T_DIM * B_DIM * Y_DIM) / 2;
  const size_t OFF_W16   = OFF_OBUF + ((size_t)T_DIM * BH) / 2;

  float* hb0 = ws + OFF_H32;            // [2][BH]
  float* hb1 = hb0 + 2 * BH;
  float* hf0 = hb1 + 2 * BH;
  float* hf1 = hf0 + 2 * BH;
  float* ystd = ws + OFF_YSTD;
  _Float16* hfr  = (_Float16*)(ws + OFF_HFRAG);   // 8 frag bufs of BH halves
  _Float16* hb0f = hfr;
  _Float16* hb1f = hfr + 2 * BH;
  _Float16* hf0f = hfr + 4 * BH;
  _Float16* hf1f = hfr + 6 * BH;
  _Float16* zfrag = (_Float16*)(ws + OFF_ZFRAG);
  _Float16* xinf  = (_Float16*)(ws + OFF_XINF);
  _Float16* zthf  = (_Float16*)(ws + OFF_ZTHF);
  _Float16* q1f   = (_Float16*)(ws + OFF_Q1F);
  _Float16* q2f   = (_Float16*)(ws + OFF_Q2F);
  _Float16* yehf  = (_Float16*)(ws + OFF_YEHF);
  _Float16* yfrag = (_Float16*)(ws + OFF_YFRG);
  _Float16* obuff = (_Float16*)(ws + OFF_OBUF);

  // ---- d_out layout ----
  const size_t TB64  = (size_t)T_DIM * B_DIM * Z_DIM;  // 4194304
  const size_t o_posm = 2;
  const size_t o_poss = 2 + TB64;
  const size_t o_prim = 2 + 2 * TB64;
  const size_t o_pris = 2 + 3 * TB64;
  const size_t o_zs   = 2 + 4 * TB64;
  const size_t o_yes  = 2 + 5 * TB64;
  const size_t o_allh = o_yes + (size_t)T_DIM * B_DIM * Y_DIM;

  // ---- init zeros (f32 hiddens, ystd, frag hiddens, z frag, accumulators) --
  hipLaunchKernelGGL(dsvm_zero_kernel, dim3((ZERO_N + 255) / 256), dim3(256),
                     0, stream, ws, (int)ZERO_N, out);
  // ---- pack y into fragments (feeds backward GRU layer 0) ----
  hipLaunchKernelGGL(dsvm_pack_y_kernel,
                     dim3((T_DIM * B_DIM * Y_DIM + 255) / 256), dim3(256),
                     0, stream, y, yfrag);

  // ---- weight swizzle (f32 -> pre-swizzled f16 fragments), bump allocator --
  _Float16* hbase = (_Float16*)(ws + OFF_W16);
  size_t hoff = 0;
  auto swz = [&](const float* src, int ld, int c0, int O, int K) -> _Float16* {
    _Float16* dst = hbase + hoff;
    hoff += (size_t)O * K;
    int total = (O / 16) * (K / 32) * 32;
    hipLaunchKernelGGL(dsvm_swizzle_kernel, dim3((total + 255) / 256),
                       dim3(256), 0, stream, src, ld, c0, O, K, dst);
    return dst;
  };
  _Float16* s_bWih0 = swz(bWih0,  32, 0, 1536,  32);
  _Float16* s_bWhh0 = swz(bWhh0, 512, 0, 1536, 512);
  _Float16* s_bWih1 = swz(bWih1, 512, 0, 1536, 512);
  _Float16* s_bWhh1 = swz(bWhh1, 512, 0, 1536, 512);
  _Float16* s_fWih0 = swz(fWih0, 128, 0, 1536, 128);
  _Float16* s_fWhh0 = swz(fWhh0, 512, 0, 1536, 512);
  _Float16* s_fWih1 = swz(fWih1, 512, 0, 1536, 512);
  _Float16* s_fWhh1 = swz(fWhh1, 512, 0, 1536, 512);
  _Float16* s_yemW  = swz(yem_W, 512, 0,   64, 512);
  _Float16* s_yemsW = swz(yem_sW, 64, 0,   32,  64);
  _Float16* s_ztrW  = swz(ztr_W,  64, 0,  128,  64);
  _Float16* s_ztrmW = swz(ztr_mW,128, 0,   64, 128);
  _Float16* s_ztrsW = swz(ztr_sW,128, 0,   64, 128);
  _Float16* s_zpW1a = swz(zp_W1, 576, 0,  128, 512);   // ob part (K=512)
  _Float16* s_zpW1b = swz(zp_W1, 576, 512, 128, 64);   // z_prev part (K=64)
  _Float16* s_zpW2  = swz(zp_W2, 128, 0,  128, 128);
  _Float16* s_zpmW  = swz(zp_mW, 128, 0,   64, 128);
  _Float16* s_zpsW  = swz(zp_sW, 128, 0,   64, 128);

  auto lin = [&](const _Float16* X1, int K1, const _Float16* W1,
                 const _Float16* X2, int K2, const _Float16* W2,
                 const float* bias, int N, int act,
                 float* o, float* o2, _Float16* of) {
    int tiles = 8 * (N / 16);
    hipLaunchKernelGGL(dsvm_linear_kernel, dim3(tiles / 4), dim3(128), 0,
                       stream, X1, K1, W1, X2, K2, W2, bias, N, act, o, o2, of);
  };
  auto gru = [&](const _Float16* Xf, int Kx, const _Float16* Hf,
                 const float* Hp, const _Float16* Wih, const _Float16* Whh,
                 const float* bih, const float* bhh,
                 float* Hn, _Float16* Hnf, float* H2, _Float16* H2f) {
    hipLaunchKernelGGL(dsvm_gru_kernel, dim3(256), dim3(192), 0, stream,
                       Xf, Kx, Hf, Hp, Wih, Whh, bih, bhh, Hn, Hnf, H2, H2f);
  };

  // ---- backward GRU scan over flipped y; store top outputs at reversed
  // index so obuff[t] == out_back_rev[t] fragments ----
  for (int i = 0; i < T_DIM; ++i) {
    int p = i & 1;
    size_t ts = (size_t)(T_DIM - 1 - i);
    gru(yfrag + ts * (B_DIM * Y_DIM), 32, hb0f + p * BH, hb0 + p * BH,
        s_bWih0, s_bWhh0, bbih0, bbhh0,
        hb0 + (p ^ 1) * BH, hb0f + (p ^ 1) * BH, nullptr, nullptr);
    gru(hb0f + (p ^ 1) * BH, 512, hb1f + p * BH, hb1 + p * BH,
        s_bWih1, s_bWhh1, bbih1, bbhh1,
        hb1 + (p ^ 1) * BH, hb1f + (p ^ 1) * BH, nullptr, obuff + ts * BH);
  }

  // ---- forward sequential scan ----
  for (int t = 0; t < T_DIM; ++t) {
    int p = t & 1;
    float* prim_t = out + o_prim + (size_t)t * B_DIM * Z_DIM;
    float* pris_t = out + o_pris + (size_t)t * B_DIM * Z_DIM;
    float* posm_t = out + o_posm + (size_t)t * B_DIM * Z_DIM;
    float* poss_t = out + o_poss + (size_t)t * B_DIM * Z_DIM;
    float* zs_t   = out + o_zs   + (size_t)t * B_DIM * Z_DIM;
    float* yes_t  = out + o_yes  + (size_t)t * B_DIM * Y_DIM;

    // prior transition
    lin(zfrag, 64, s_ztrW, nullptr, 0, nullptr, ztr_b, 128, 1,
        nullptr, nullptr, zthf);
    lin(zthf, 128, s_ztrmW, nullptr, 0, nullptr, ztr_mb, 64, 0,
        prim_t, nullptr, nullptr);
    lin(zthf, 128, s_ztrsW, nullptr, 0, nullptr, ztr_sb, 64, 2,
        pris_t, nullptr, nullptr);
    // posterior (cat(ob, z_prev) as two-input GEMM)
    lin(obuff + (size_t)t * BH, 512, s_zpW1a, zfrag, 64, s_zpW1b,
        zp_b1, 128, 1, nullptr, nullptr, q1f);
    lin(q1f, 128, s_zpW2, nullptr, 0, nullptr, zp_b2, 128, 1,
        nullptr, nullptr, q2f);
    lin(q2f, 128, s_zpmW, nullptr, 0, nullptr, zp_mb, 64, 0,
        posm_t, nullptr, nullptr);
    lin(q2f, 128, s_zpsW, nullptr, 0, nullptr, zp_sb, 64, 2,
        poss_t, nullptr, nullptr);
    // sample z_t, assemble x_in fragments, accumulate KLD
    hipLaunchKernelGGL(dsvm_sample_kld_kernel, dim3(32), dim3(256), 0, stream,
                       eps + (size_t)t * B_DIM * Z_DIM, posm_t, poss_t, prim_t,
                       pris_t,
                       (t > 0) ? (y + (size_t)(t - 1) * B_DIM * Y_DIM) : nullptr,
                       ystd, zs_t, zfrag, xinf, out /* kld at out[0] */);
    // forward GRU (writes all_h)
    gru(xinf, 128, hf0f + p * BH, hf0 + p * BH, s_fWih0, s_fWhh0,
        fbih0, fbhh0, hf0 + (p ^ 1) * BH, hf0f + (p ^ 1) * BH,
        out + o_allh + (size_t)(2 * t) * BH, nullptr);
    gru(hf0f + (p ^ 1) * BH, 512, hf1f + p * BH, hf1 + p * BH,
        s_fWih1, s_fWhh1, fbih1, fbhh1,
        hf1 + (p ^ 1) * BH, hf1f + (p ^ 1) * BH,
        out + o_allh + (size_t)(2 * t + 1) * BH, nullptr);
    // emission std + NLL
    lin(hf1f + (p ^ 1) * BH, 512, s_yemW, nullptr, 0, nullptr, yem_b, 64, 1,
        nullptr, nullptr, yehf);
    lin(yehf, 64, s_yemsW, nullptr, 0, nullptr, yem_sb, 32, 2,
        yes_t, ystd, nullptr);
    hipLaunchKernelGGL(dsvm_nll_kernel, dim3(16), dim3(256), 0, stream,
                       y + (size_t)t * B_DIM * Y_DIM, ystd, out + 1);
  }
}
